// Attention_29394756174346
// MI455X (gfx1250) — matrix operand order repaired and checked by reading, not compiled, not run on hardware
//
#include <hip/hip_runtime.h>

// ---------------- types ----------------
typedef __bf16 bf16_t;
typedef __bf16 bf16x8 __attribute__((ext_vector_type(8)));   // 16 bytes
typedef __bf16 v16bf  __attribute__((ext_vector_type(16)));  // 32 bytes (8 VGPRs)
typedef float  v8f    __attribute__((ext_vector_type(8)));

// ---------------- problem constants ----------------
constexpr int Bc   = 2;
constexpr int Sc   = 2048;
constexpr int HIDc = 2048;
constexpr int Hc   = 16;
constexpr int Gc   = 4;
constexpr int Dc   = 128;
constexpr int REP  = Hc / Gc;          // 4
constexpr int Mrows = Bc * Sc;         // 4096 tokens
constexpr int NQ   = Hc * Dc;          // 2048
constexpr int NKV  = Gc * Dc;          // 512
constexpr float NEGF  = -1e30f;
constexpr float SCALE = 0.08838834764831845f;  // 1/sqrt(128)

// ---------------- fragment helpers ----------------
__device__ inline v16bf frag_combine(bf16x8 lo8, bf16x8 hi8) {
  v16bf r;
#pragma unroll
  for (int i = 0; i < 8; ++i) { r[i] = lo8[i]; r[i + 8] = hi8[i]; }
  return r;
}

// A-matrix (16x32, row-major source, ldK elements): lane holds row (lane&15),
// k = (lane>>4)*16 + e, e=0..15 contiguous -> two b128 loads at +0 / +8.
__device__ inline v16bf load_a_frag(const bf16_t* p /* row base + kbase */) {
  return frag_combine(*(const bf16x8*)p, *(const bf16x8*)(p + 8));
}

// B-matrix (32x16): lane holds col (lane&15), k = (lane>>4)*16 + e, e=0..15
// contiguous -> two b128 loads at +0 / +8.
__device__ inline v16bf load_b_frag(const bf16_t* p /* col base + koff */) {
  return frag_combine(*(const bf16x8*)p, *(const bf16x8*)(p + 8));
}

__device__ inline v8f wmma_bf16(v16bf a, v16bf b, v8f c) {
  return __builtin_amdgcn_wmma_f32_16x16x32_bf16(
      false, a, false, b, (short)0, c, false, false);
}

// ---------------- elementwise kernels ----------------
__global__ __launch_bounds__(256)
void k_f32_to_bf16(const float* __restrict__ src, bf16_t* __restrict__ dst, int n) {
  int i = blockIdx.x * blockDim.x + threadIdx.x;
  if (i < n) dst[i] = (bf16_t)src[i];
}

// W [K,N] fp32 -> Wt [N,K] bf16
__global__ __launch_bounds__(256)
void k_transpose_w(const float* __restrict__ W, bf16_t* __restrict__ Wt, int K, int N) {
  int i = blockIdx.x * blockDim.x + threadIdx.x;
  if (i >= K * N) return;
  int k = i / N, n = i - k * N;
  Wt[(size_t)n * K + k] = (bf16_t)W[i];
}

// RoPE: src f32 [B,S,NH,D] -> dst bf16 same layout. cos/sin: [S,D]
__global__ __launch_bounds__(256)
void k_rope(const float* __restrict__ src, const float* __restrict__ cosT,
            const float* __restrict__ sinT, bf16_t* __restrict__ dst, int NH) {
  int i = blockIdx.x * blockDim.x + threadIdx.x;
  int total = Bc * Sc * NH * Dc;
  if (i >= total) return;
  int d = i & (Dc - 1);
  int s = (i / (NH * Dc)) % Sc;
  float v = src[i];
  float rot = (d < Dc / 2) ? -src[i + Dc / 2] : src[i - Dc / 2];
  float c = cosT[s * Dc + d], sn = sinT[s * Dc + d];
  dst[i] = (bf16_t)(v * c + rot * sn);
}

// V f32 [B,S,G,D] -> Vt bf16 [B,G,D,S]
__global__ __launch_bounds__(256)
void k_transpose_v(const float* __restrict__ V, bf16_t* __restrict__ Vt) {
  int i = blockIdx.x * blockDim.x + threadIdx.x;
  int total = Bc * Sc * Gc * Dc;
  if (i >= total) return;
  int d = i & (Dc - 1);
  int g = (i / Dc) & (Gc - 1);
  int s = (i / (Dc * Gc)) % Sc;
  int b = i / (Dc * Gc * Sc);
  Vt[(((size_t)b * Gc + g) * Dc + d) * Sc + s] = (bf16_t)V[i];
}

// ---------------- GEMM: C[M,N] f32 = A[M,K] bf16 x Bt[N,K] bf16 ----------------
// 256 threads = 8 waves as 4(M) x 2(N); wave tile 32x32; block tile 128x64.
__global__ __launch_bounds__(256)
void k_gemm_bf16(const bf16_t* __restrict__ A, const bf16_t* __restrict__ Bt,
                 float* __restrict__ C, int M, int N, int K) {
  const int lane = threadIdx.x & 31;
  const int wave = threadIdx.x >> 5;
  const int ntiles = N >> 6;
  const int bm = (blockIdx.x / ntiles) << 7;
  const int bn = (blockIdx.x % ntiles) << 6;
  const int wm = bm + (wave & 3) * 32;
  const int wn = bn + (wave >> 2) * 32;
  const int lo = lane & 15, hi = lane >> 4;

  v8f c00 = {}, c01 = {}, c10 = {}, c11 = {};

  const bf16_t* a0p = A  + (size_t)(wm + lo)      * K + hi * 16;
  const bf16_t* a1p = A  + (size_t)(wm + 16 + lo) * K + hi * 16;
  const bf16_t* b0p = Bt + (size_t)(wn + lo)      * K + hi * 16;
  const bf16_t* b1p = Bt + (size_t)(wn + 16 + lo) * K + hi * 16;

  for (int k0 = 0; k0 < K; k0 += 32) {
    __builtin_prefetch(a0p + k0 + 128, 0, 1);   // global_prefetch_b8
    __builtin_prefetch(b0p + k0 + 128, 0, 1);
    v16bf a0 = load_a_frag(a0p + k0);
    v16bf a1 = load_a_frag(a1p + k0);
    v16bf b0 = load_b_frag(b0p + k0);
    v16bf b1 = load_b_frag(b1p + k0);
    c00 = wmma_bf16(a0, b0, c00);
    c01 = wmma_bf16(a0, b1, c01);
    c10 = wmma_bf16(a1, b0, c10);
    c11 = wmma_bf16(a1, b1, c11);
  }

#pragma unroll
  for (int r = 0; r < 8; ++r) {
    int m0 = wm + hi * 8 + r;
    C[(size_t)m0 * N + wn + lo]             = c00[r];
    C[(size_t)m0 * N + wn + 16 + lo]        = c01[r];
    C[(size_t)(m0 + 16) * N + wn + lo]      = c10[r];
    C[(size_t)(m0 + 16) * N + wn + 16 + lo] = c11[r];
  }
}

// ---------------- flash attention ----------------
// One wave per (b, h, 16-query tile). Online softmax over 32-key blocks.
__global__ __launch_bounds__(128)
void k_attn(const bf16_t* __restrict__ Q,   // [B,S,H,D] bf16 (post-RoPE)
            const bf16_t* __restrict__ Kx,  // [B,S,G,D] bf16 (post-RoPE)
            const bf16_t* __restrict__ Vt,  // [B,G,D,S] bf16
            bf16_t* __restrict__ AO) {      // [B,S,H*D] bf16
  __shared__ __align__(16) bf16_t plds[4][16 * 32];

  const int lane = threadIdx.x & 31;
  const int wib  = threadIdx.x >> 5;
  const int wid  = blockIdx.x * 4 + wib;
  const int QT   = Sc / 16;                  // 128 query tiles
  const int qt   = wid % QT;
  const int h    = (wid / QT) % Hc;
  const int b    = wid / (QT * Hc);
  const int g    = h / REP;
  const int lo = lane & 15, hi = lane >> 4;

  // Q A-fragments: 4 k-steps over D=128
  v16bf qf[4];
  {
    const bf16_t* qrow = Q + (((size_t)b * Sc + qt * 16 + lo) * Hc + h) * Dc + hi * 16;
#pragma unroll
    for (int kk = 0; kk < 4; ++kk) qf[kk] = load_a_frag(qrow + kk * 32);
  }

  float rm[8], rl[8];
  v8f o[8];
#pragma unroll
  for (int r = 0; r < 8; ++r) { rm[r] = NEGF; rl[r] = 0.0f; }
#pragma unroll
  for (int dt = 0; dt < 8; ++dt) o[dt] = (v8f){};

  bf16_t* myp = plds[wib];
  const int kend = qt * 16 + 16;             // causal horizon
  const int qbase = qt * 16 + hi * 8;

  for (int kb0 = 0; kb0 < kend; kb0 += 32) {
    // ---- S = Q K^T : two 16-key column tiles ----
    v8f s0 = {}, s1 = {};
    const bf16_t* k0row = Kx + (((size_t)b * Sc + kb0 + lo) * Gc + g) * Dc + hi * 16;
    const bf16_t* k1row = k0row + (size_t)16 * Gc * Dc;
#pragma unroll
    for (int kk = 0; kk < 4; ++kk) {
      v16bf kf0 = load_b_frag(k0row + kk * 32);
      v16bf kf1 = load_b_frag(k1row + kk * 32);
      s0 = wmma_bf16(qf[kk], kf0, s0);
      s1 = wmma_bf16(qf[kk], kf1, s1);
    }

    // ---- scale + causal mask + online softmax ----
    float sm0[8], sm1[8], mnew[8], alpha[8];
#pragma unroll
    for (int r = 0; r < 8; ++r) {
      int qi = qbase + r;
      float x0 = s0[r] * SCALE;
      float x1 = s1[r] * SCALE;
      if (kb0 + lo > qi)      x0 = NEGF;
      if (kb0 + 16 + lo > qi) x1 = NEGF;
      sm0[r] = x0; sm1[r] = x1;
    }
#pragma unroll
    for (int r = 0; r < 8; ++r) {
      float mx = fmaxf(sm0[r], sm1[r]);
#pragma unroll
      for (int off = 1; off < 16; off <<= 1)
        mx = fmaxf(mx, __shfl_xor(mx, off, 32));
      mnew[r]  = fmaxf(rm[r], mx);
      alpha[r] = __expf(rm[r] - mnew[r]);
      rm[r]    = mnew[r];
    }
#pragma unroll
    for (int r = 0; r < 8; ++r) {
      float p0 = __expf(sm0[r] - mnew[r]);
      float p1 = __expf(sm1[r] - mnew[r]);
      int row = hi * 8 + r;
      myp[row * 32 + lo]      = (bf16_t)p0;
      myp[row * 32 + 16 + lo] = (bf16_t)p1;
      float ps = p0 + p1;
#pragma unroll
      for (int off = 1; off < 16; off <<= 1)
        ps += __shfl_xor(ps, off, 32);
      rl[r] = rl[r] * alpha[r] + ps;
    }
#pragma unroll
    for (int dt = 0; dt < 8; ++dt)
#pragma unroll
      for (int r = 0; r < 8; ++r) o[dt][r] *= alpha[r];

    // LDS store -> load re-layout (C-fragment -> A-fragment); same-wave only
    asm volatile("s_wait_dscnt 0" ::: "memory");
    v16bf pf = load_a_frag(myp + lo * 32 + hi * 16);

    // ---- O += P V ----
#pragma unroll
    for (int dt = 0; dt < 8; ++dt) {
      const bf16_t* vp =
          Vt + (((size_t)b * Gc + g) * Dc + dt * 16 + lo) * Sc + kb0 + hi * 16;
      v16bf vf = load_b_frag(vp);
      o[dt] = wmma_bf16(pf, vf, o[dt]);
    }
  }

  // ---- normalize + store ----
#pragma unroll
  for (int dt = 0; dt < 8; ++dt) {
#pragma unroll
    for (int r = 0; r < 8; ++r) {
      int qi = qt * 16 + hi * 8 + r;
      int d  = dt * 16 + lo;
      AO[((size_t)b * Sc + qi) * (Hc * Dc) + h * Dc + d] =
          (bf16_t)(o[dt][r] / rl[r]);
    }
  }
}

// ---------------- host ----------------
extern "C" void kernel_launch(void* const* d_in, const int* in_sizes, int n_in,
                              void* d_out, int out_size, void* d_ws, size_t ws_size,
                              hipStream_t stream) {
  const float* x    = (const float*)d_in[0];
  const float* cosT = (const float*)d_in[1];
  const float* sinT = (const float*)d_in[2];
  const float* Wq   = (const float*)d_in[3];
  const float* Wk   = (const float*)d_in[4];
  const float* Wv   = (const float*)d_in[5];
  const float* Wo   = (const float*)d_in[6];
  float* out = (float*)d_out;

  char* p = (char*)d_ws;
  auto carve = [&](size_t bytes) -> void* {
    char* r = p;
    p += (bytes + 255) & ~(size_t)255;
    return (void*)r;
  };
  bf16_t* xb  = (bf16_t*)carve((size_t)Mrows * HIDc * 2);
  bf16_t* wqt = (bf16_t*)carve((size_t)NQ   * HIDc * 2);
  bf16_t* wkt = (bf16_t*)carve((size_t)NKV  * HIDc * 2);
  bf16_t* wvt = (bf16_t*)carve((size_t)NKV  * HIDc * 2);
  bf16_t* wot = (bf16_t*)carve((size_t)HIDc * NQ   * 2);
  float*  q32 = (float*) carve((size_t)Mrows * NQ  * 4);
  float*  k32 = (float*) carve((size_t)Mrows * NKV * 4);
  float*  v32 = (float*) carve((size_t)Mrows * NKV * 4);
  bf16_t* qb  = (bf16_t*)carve((size_t)Mrows * NQ  * 2);
  bf16_t* kbf = (bf16_t*)carve((size_t)Mrows * NKV * 2);
  bf16_t* vt  = (bf16_t*)carve((size_t)Mrows * NKV * 2);
  bf16_t* ao  = (bf16_t*)q32;   // overlay: q32 dead after RoPE

  const int T = 256;
  auto blocks = [](long long n, int t) { return (int)((n + t - 1) / t); };

  // 1) convert x, transpose+convert weights
  k_f32_to_bf16<<<blocks((long long)Mrows * HIDc, T), T, 0, stream>>>(x, xb, Mrows * HIDc);
  k_transpose_w<<<blocks((long long)HIDc * NQ,  T), T, 0, stream>>>(Wq, wqt, HIDc, NQ);
  k_transpose_w<<<blocks((long long)HIDc * NKV, T), T, 0, stream>>>(Wk, wkt, HIDc, NKV);
  k_transpose_w<<<blocks((long long)HIDc * NKV, T), T, 0, stream>>>(Wv, wvt, HIDc, NKV);
  k_transpose_w<<<blocks((long long)NQ * HIDc,  T), T, 0, stream>>>(Wo, wot, NQ, HIDc);

  // 2) QKV projection GEMMs (WMMA)
  k_gemm_bf16<<<(Mrows / 128) * (NQ / 64),  T, 0, stream>>>(xb, wqt, q32, Mrows, NQ,  HIDc);
  k_gemm_bf16<<<(Mrows / 128) * (NKV / 64), T, 0, stream>>>(xb, wkt, k32, Mrows, NKV, HIDc);
  k_gemm_bf16<<<(Mrows / 128) * (NKV / 64), T, 0, stream>>>(xb, wvt, v32, Mrows, NKV, HIDc);

  // 3) RoPE (Q,K) + V transpose, all to bf16
  k_rope<<<blocks((long long)Mrows * NQ,  T), T, 0, stream>>>(q32, cosT, sinT, qb,  Hc);
  k_rope<<<blocks((long long)Mrows * NKV, T), T, 0, stream>>>(k32, cosT, sinT, kbf, Gc);
  k_transpose_v<<<blocks((long long)Mrows * NKV, T), T, 0, stream>>>(v32, vt);

  // 4) causal flash attention (WMMA)  — 4096 waves
  k_attn<<<(Bc * Hc * (Sc / 16)) / 4, 128, 0, stream>>>(qb, kbf, vt, ao);

  // 5) output projection (WMMA) -> f32 d_out
  k_gemm_bf16<<<(Mrows / 128) * (HIDc / 64), T, 0, stream>>>(ao, wot, out, Mrows, HIDc, NQ);
}
